// NeRF_90220083020073
// MI455X (gfx1250) — matrix operand order repaired and checked by reading, not compiled, not run on hardware
//
#include <hip/hip_runtime.h>
#include <hip/hip_bf16.h>

// Fused instant-NGP NeRF forward for gfx1250 (MI455X).
//  - hash-grid encode: scalar gathers (64MB tables are L2-resident)
//  - all MLPs: v_wmma_f32_16x16x32_f16, activations staged in LDS as f16
//  - one kernel, no HBM round-trips for intermediates

typedef __attribute__((ext_vector_type(16))) _Float16 v16h;
typedef __attribute__((ext_vector_type(8)))  float    v8f;

#define K_LEVELS 16
#define K_T      (1u << 19)
#define K_NPB    128   /* points per block */
#define K_TPB    128   /* threads per block (4 wave32) */

__constant__ int c_res[K_LEVELS] = {16, 22, 30, 42, 58, 80, 111, 153,
                                    212, 294, 406, 561, 776, 1072, 1482, 2048};

__device__ __forceinline__ int imin(int a, int b) { return a < b ? a : b; }
__device__ __forceinline__ int imax(int a, int b) { return a > b ? a : b; }

__device__ __forceinline__ v8f wmma32(v16h a, v16h b, v8f c) {
  // D = A(16x32 f16) * B(32x16 f16) + C(16x16 f32)
  return __builtin_amdgcn_wmma_f32_16x16x32_f16(false, a, false, b,
                                                (short)0, c, false, false);
}

// A fragment (16x32, f16) from row-major LDS [row][stride] at column base kbase.
// ISA layout: lane L (L<16) holds M=L, K in {h*8..h*8+7, 16+h*8..16+h*8+7}, h=L>>4.
__device__ __forceinline__ v16h lds_loadA(const _Float16* base, int row0,
                                          int stride, int kbase) {
  const int lane = (int)(threadIdx.x & 31u);
  const int m  = lane & 15;
  const int hh = lane >> 4;
  const _Float16* r = base + (row0 + m) * stride + kbase + hh * 8;
  v16h a;
#pragma unroll
  for (int i = 0; i < 8; ++i) a[i] = r[i];
#pragma unroll
  for (int i = 0; i < 8; ++i) a[8 + i] = r[16 + i];
  return a;
}

// B fragment (32x16, f16) from row-major LDS weight [K][nfull]:
// lane L holds column N=n0+(L&15), K in {h*8..h*8+7, 16+h*8..16+h*8+7}, h=L>>4.
__device__ __forceinline__ v16h lds_loadB(const _Float16* wt, int nfull,
                                          int kbase, int n0) {
  const int lane = (int)(threadIdx.x & 31u);
  const int nn = lane & 15;
  const int hh = lane >> 4;
  const _Float16* r = wt + (kbase + hh * 8) * nfull + n0 + nn;
  v16h b;
#pragma unroll
  for (int i = 0; i < 8; ++i) b[i] = r[i * nfull];
#pragma unroll
  for (int i = 0; i < 8; ++i) b[8 + i] = r[(16 + i) * nfull];
  return b;
}

__global__ __launch_bounds__(K_TPB) void NeRF_90220083020073_kernel(
    const float* __restrict__ x, const float* __restrict__ d,
    const float* __restrict__ tables,
    const float* __restrict__ w1,  const float* __restrict__ b1,
    const float* __restrict__ w2,  const float* __restrict__ b2,
    const float* __restrict__ wd1, const float* __restrict__ bd1,
    const float* __restrict__ wd2, const float* __restrict__ bd2,
    const float* __restrict__ wd3, const float* __restrict__ bd3,
    const float* __restrict__ xyzmin, const float* __restrict__ xyzmax,
    float* __restrict__ out_sigma, float* __restrict__ out_rgb, int npts) {

  // ---- LDS ------------------------------------------------------------
  __shared__ _Float16 sW1[32 * 64];    // w1  [K=32][N=64]
  __shared__ _Float16 sW2[64 * 32];    // w2  [K=64][N=32] (17 padded to 32)
  __shared__ _Float16 sWd1[32 * 64];   // wd1 [K=32][N=64]
  __shared__ _Float16 sWd2[64 * 64];   // wd2 [K=64][N=64]
  __shared__ _Float16 sWd3[64 * 16];   // wd3 [K=64][N=16] (3 padded to 16)
  __shared__ float    sBias[240];      // b1@0, b2@64, bd1@96, bd2@160, bd3@224
  __shared__ _Float16 sEnc[K_NPB * 32];  // hash-grid features, row-major
  __shared__ _Float16 sCat[K_NPB * 64];  // [sh(16) | feat(16)] then wd2 output
  __shared__ _Float16 sBuf[K_NPB * 64];  // hidden-layer scratch

  const int tid = (int)threadIdx.x;

  // ---- stage weights (f32 -> f16) & biases ---------------------------
  for (int i = tid; i < 32 * 64; i += K_TPB) sW1[i]  = (_Float16)w1[i];
  for (int i = tid; i < 64 * 32; i += K_TPB) {
    int r = i >> 5, c = i & 31;
    sW2[i] = (_Float16)((c < 17) ? w2[r * 17 + c] : 0.f);
  }
  for (int i = tid; i < 32 * 64; i += K_TPB) sWd1[i] = (_Float16)wd1[i];
  for (int i = tid; i < 64 * 64; i += K_TPB) sWd2[i] = (_Float16)wd2[i];
  for (int i = tid; i < 64 * 16; i += K_TPB) {
    int r = i >> 4, c = i & 15;
    sWd3[i] = (_Float16)((c < 3) ? wd3[r * 3 + c] : 0.f);
  }
  for (int i = tid; i < 64; i += K_TPB) sBias[i]       = b1[i];
  for (int i = tid; i < 32; i += K_TPB) sBias[64 + i]  = (i < 17) ? b2[i] : 0.f;
  for (int i = tid; i < 64; i += K_TPB) sBias[96 + i]  = bd1[i];
  for (int i = tid; i < 64; i += K_TPB) sBias[160 + i] = bd2[i];
  for (int i = tid; i < 16; i += K_TPB) sBias[224 + i] = (i < 3) ? bd3[i] : 0.f;

  // ---- phase 1: per-thread hash-grid encode + SH ---------------------
  const int pAbs = (int)blockIdx.x * K_NPB + tid;
  const int p = pAbs < npts ? pAbs : npts - 1;   // clamp reads, keep EXEC full

  const float x0 = x[p * 3 + 0], x1 = x[p * 3 + 1], x2 = x[p * 3 + 2];
  const float mn0 = xyzmin[0], mn1 = xyzmin[1], mn2 = xyzmin[2];
  const float mx0 = xyzmax[0], mx1 = xyzmax[1], mx2 = xyzmax[2];
  const float xc0 = ((x0 - mn0) / (mx0 - mn0) * 2.f - 1.f) * 0.25f + 0.5f;
  const float xc1 = ((x1 - mn1) / (mx1 - mn1) * 2.f - 1.f) * 0.25f + 0.5f;
  const float xc2 = ((x2 - mn2) / (mx2 - mn2) * 2.f - 1.f) * 0.25f + 0.5f;

#pragma unroll 1
  for (int l = 0; l < K_LEVELS; ++l) {
    const int res = c_res[l];
    const float rf = (float)res;
    const float px = xc0 * rf, py = xc1 * rf, pz = xc2 * rf;
    const float fx = floorf(px), fy = floorf(py), fz = floorf(pz);
    const float wx = px - fx, wy = py - fy, wz = pz - fz;
    const int ix = (int)fx, iy = (int)fy, iz = (int)fz;
    const int r1 = res + 1;
    const bool dense = (long long)r1 * r1 * r1 <= (long long)K_T;
    const float2* tb = (const float2*)tables + (size_t)l * K_T;
    float a0 = 0.f, a1 = 0.f;
#pragma unroll
    for (int c = 0; c < 8; ++c) {
      const int ox = c & 1, oy = (c >> 1) & 1, oz = (c >> 2) & 1;
      const int cx = ix + ox, cy = iy + oy, cz = iz + oz;
      const float wt = (ox ? wx : 1.f - wx) * (oy ? wy : 1.f - wy) *
                       (oz ? wz : 1.f - wz);
      unsigned idx;
      if (dense) {
        const int dx = imin(imax(cx, 0), res);
        const int dy = imin(imax(cy, 0), res);
        const int dz = imin(imax(cz, 0), res);
        idx = (unsigned)(dx + dy * r1 + dz * r1 * r1);
      } else {
        const unsigned h = (unsigned)cx * 1u ^ (unsigned)cy * 2654435761u ^
                           (unsigned)cz * 805459861u;
        idx = h & (K_T - 1u);
      }
      const float2 f = tb[idx];     // L2-resident gather
      a0 += wt * f.x;
      a1 += wt * f.y;
    }
    sEnc[tid * 32 + 2 * l]     = (_Float16)a0;
    sEnc[tid * 32 + 2 * l + 1] = (_Float16)a1;
  }

  // SH degree 4 (tcnn constants) -> sCat cols 0..15
  {
    const float dx = d[p * 3 + 0], dy = d[p * 3 + 1], dz = d[p * 3 + 2];
    const float xx = dx * dx, yy = dy * dy, zz = dz * dz;
    const float xy = dx * dy, yz = dy * dz, xz = dx * dz;
    float sh[16];
    sh[0]  = 0.28209479177387814f;
    sh[1]  = -0.48860251190291987f * dy;
    sh[2]  = 0.48860251190291987f * dz;
    sh[3]  = -0.48860251190291987f * dx;
    sh[4]  = 1.0925484305920792f * xy;
    sh[5]  = -1.0925484305920792f * yz;
    sh[6]  = 0.94617469575756f * zz - 0.31539156525252f;
    sh[7]  = -1.0925484305920792f * xz;
    sh[8]  = 0.5462742152960396f * (xx - yy);
    sh[9]  = 0.5900435899266435f * dy * (-3.f * xx + yy);
    sh[10] = 2.890611442640554f * xy * dz;
    sh[11] = 0.4570457994644657f * dy * (1.f - 5.f * zz);
    sh[12] = 0.3731763325901154f * dz * (5.f * zz - 3.f);
    sh[13] = 0.4570457994644657f * dx * (1.f - 5.f * zz);
    sh[14] = 1.445305721320277f * dz * (xx - yy);
    sh[15] = 0.5900435899266435f * dx * (-xx + 3.f * yy);
#pragma unroll
    for (int i = 0; i < 16; ++i) sCat[tid * 64 + i] = (_Float16)sh[i];
  }

  __syncthreads();

  // ---- phase 2: WMMA MLPs, one wave per 32 points (2 tiles of 16) ----
  const int lane = tid & 31;
  const int wave = tid >> 5;
  const int n  = lane & 15;   // C/D column within 16-wide block
  const int hh = lane >> 4;   // C/D row-half select

#pragma unroll 1
  for (int tile = 0; tile < 2; ++tile) {
    const int row0 = wave * 32 + tile * 16;

    // --- density layer 1: enc[16x32] @ w1[32x64], +b1, ReLU -> sBuf ---
    {
      const v16h a = lds_loadA(sEnc, row0, 32, 0);
#pragma unroll
      for (int j = 0; j < 4; ++j) {
        v8f c = {};
        c = wmma32(a, lds_loadB(sW1, 64, 0, j * 16), c);
#pragma unroll
        for (int r = 0; r < 8; ++r) {
          const int m = r + hh * 8;
          float v = c[r] + sBias[0 + j * 16 + n];
          v = fmaxf(v, 0.f);
          sBuf[(row0 + m) * 64 + j * 16 + n] = (_Float16)v;
        }
      }
    }

    // --- density layer 2: h1[16x64] @ w2[64x32pad], sigma + feat ------
    {
      const v16h a0 = lds_loadA(sBuf, row0, 64, 0);
      const v16h a1 = lds_loadA(sBuf, row0, 64, 32);
#pragma unroll
      for (int j = 0; j < 2; ++j) {
        v8f c = {};
        c = wmma32(a0, lds_loadB(sW2, 32, 0, j * 16), c);
        c = wmma32(a1, lds_loadB(sW2, 32, 32, j * 16), c);
        const int N = j * 16 + n;
#pragma unroll
        for (int r = 0; r < 8; ++r) {
          const int m = r + hh * 8;
          const float v = c[r] + sBias[64 + N];
          if (N == 0) {
            const int prow = (int)blockIdx.x * K_NPB + row0 + m;
            const float s = v > 20.f ? v : log1pf(expf(v));  // softplus
            if (prow < npts) out_sigma[prow] = s;
          } else if (N <= 16) {
            // feat[:, N-1] -> concat columns 16..31
            sCat[(row0 + m) * 64 + 15 + N] = (_Float16)v;
          }
        }
      }
    }

    // --- color layer 1: [sh|feat][16x32] @ wd1[32x64], ReLU -> sBuf ---
    {
      const v16h a = lds_loadA(sCat, row0, 64, 0);
#pragma unroll
      for (int j = 0; j < 4; ++j) {
        v8f c = {};
        c = wmma32(a, lds_loadB(sWd1, 64, 0, j * 16), c);
#pragma unroll
        for (int r = 0; r < 8; ++r) {
          const int m = r + hh * 8;
          float v = c[r] + sBias[96 + j * 16 + n];
          v = fmaxf(v, 0.f);
          sBuf[(row0 + m) * 64 + j * 16 + n] = (_Float16)v;
        }
      }
    }

    // --- color layer 2: h[16x64] @ wd2[64x64], ReLU -> sCat -----------
    {
      const v16h a0 = lds_loadA(sBuf, row0, 64, 0);
      const v16h a1 = lds_loadA(sBuf, row0, 64, 32);
#pragma unroll
      for (int j = 0; j < 4; ++j) {
        v8f c = {};
        c = wmma32(a0, lds_loadB(sWd2, 64, 0, j * 16), c);
        c = wmma32(a1, lds_loadB(sWd2, 64, 32, j * 16), c);
#pragma unroll
        for (int r = 0; r < 8; ++r) {
          const int m = r + hh * 8;
          float v = c[r] + sBias[160 + j * 16 + n];
          v = fmaxf(v, 0.f);
          sCat[(row0 + m) * 64 + j * 16 + n] = (_Float16)v;
        }
      }
    }

    // --- color layer 3: h[16x64] @ wd3[64x16pad], sigmoid -> rgb ------
    {
      const v16h a0 = lds_loadA(sCat, row0, 64, 0);
      const v16h a1 = lds_loadA(sCat, row0, 64, 32);
      v8f c = {};
      c = wmma32(a0, lds_loadB(sWd3, 16, 0, 0), c);
      c = wmma32(a1, lds_loadB(sWd3, 16, 32, 0), c);
#pragma unroll
      for (int r = 0; r < 8; ++r) {
        const int m = r + hh * 8;
        const int prow = (int)blockIdx.x * K_NPB + row0 + m;
        if (n < 3 && prow < npts) {
          const float v = c[r] + sBias[224 + n];
          out_rgb[prow * 3 + n] = 1.f / (1.f + expf(-v));
        }
      }
    }
  }
}

extern "C" void kernel_launch(void* const* d_in, const int* in_sizes, int n_in,
                              void* d_out, int out_size, void* d_ws,
                              size_t ws_size, hipStream_t stream) {
  (void)n_in; (void)out_size; (void)d_ws; (void)ws_size;
  const float* x      = (const float*)d_in[0];
  const float* d      = (const float*)d_in[1];
  const float* tables = (const float*)d_in[2];
  const float* w1     = (const float*)d_in[3];
  const float* b1     = (const float*)d_in[4];
  const float* w2     = (const float*)d_in[5];
  const float* b2     = (const float*)d_in[6];
  const float* wd1    = (const float*)d_in[7];
  const float* bd1    = (const float*)d_in[8];
  const float* wd2    = (const float*)d_in[9];
  const float* bd2    = (const float*)d_in[10];
  const float* wd3    = (const float*)d_in[11];
  const float* bd3    = (const float*)d_in[12];
  const float* xyzmin = (const float*)d_in[13];
  const float* xyzmax = (const float*)d_in[14];

  const int npts = in_sizes[0] / 3;
  float* out_sigma = (float*)d_out;
  float* out_rgb   = out_sigma + npts;

  const int blocks = (npts + K_NPB - 1) / K_NPB;
  NeRF_90220083020073_kernel<<<blocks, K_TPB, 0, stream>>>(
      x, d, tables, w1, b1, w2, b2, wd1, bd1, wd2, bd2, wd3, bd3,
      xyzmin, xyzmax, out_sigma, out_rgb, npts);
}
